// Relative_Attention_39676907885434
// MI455X (gfx1250) — compile-verified, matrix operand order checked by reading, not run on hardware
//
#include <hip/hip_runtime.h>
#include <hip/hip_bf16.h>
#include <stdint.h>

// ---------------------------------------------------------------------------
// Types for CDNA5 WMMA bf16 fragments
// ---------------------------------------------------------------------------
typedef __bf16 bf16_t;
typedef __attribute__((ext_vector_type(16))) __bf16 v16bf;
typedef __attribute__((ext_vector_type(8)))  __bf16 bf16x8;
typedef __attribute__((ext_vector_type(8)))  float  v8f;

union Frag16 {
    v16bf  v;
    bf16x8 h[2];
};

// ---------------------------------------------------------------------------
// fp32 -> bf16 elementwise convert
// ---------------------------------------------------------------------------
__global__ __launch_bounds__(256)
void cvt_f32_to_bf16(const float* __restrict__ in, bf16_t* __restrict__ out, long n)
{
    long i = (long)blockIdx.x * blockDim.x + threadIdx.x;
    long stride = (long)gridDim.x * blockDim.x;
    for (; i < n; i += stride)
        out[i] = (bf16_t)in[i];
}

// ---------------------------------------------------------------------------
// LDS-staged, double-buffered, batched bf16 WMMA GEMM:
//   C[z][m,n] = sum_k A[z/divA][m,k] * B[z%modB][n,k]  (+ bias[n])
//
// Block tile: 128(M) x 128(N), K staged in slabs of 64 through LDS.
// 8 waves (256 threads): wave = (waveM in 0..1) x (waveN in 0..3);
// each wave owns a 64x32 register tile = 4 A-frags x 2 B-frags x 8 acc,
// 16 v_wmma_f32_16x16x32_bf16 per K-slab.
// Global -> LDS staging uses GLOBAL_LOAD_ASYNC_TO_LDS_B128 (ASYNCcnt),
// double-buffered so the next slab streams while the current one computes.
// ---------------------------------------------------------------------------
#define KSLAB      64
#define LDSROW     72            // 64 + 8 bf16 pad -> 144B row, breaks bank conflicts
#define TILE_MN    128

__global__ __launch_bounds__(256)
void gemm_bf16_wmma(const bf16_t* __restrict__ A, long bsA, int divA, long ldA,
                    const bf16_t* __restrict__ B, long bsB, int modB, long ldB,
                    void* __restrict__ C, long bsC, long sCm, long sCn, int c_is_bf16,
                    const float* __restrict__ bias,
                    int M, int N, int K)
{
    __shared__ bf16_t smA[2][TILE_MN * LDSROW];
    __shared__ bf16_t smB[2][TILE_MN * LDSROW];

    const int lane  = threadIdx.x;                 // 0..31 (wave32)
    const int wave  = threadIdx.y;                 // 0..7
    const int tid   = wave * 32 + lane;            // 0..255
    const int z     = blockIdx.z;

    const int tileM = blockIdx.y * TILE_MN;
    const int tileN = blockIdx.x * TILE_MN;

    const bf16_t* Ab = A + (long)(z / divA) * bsA;
    const bf16_t* Bb = B + (long)(z % modB) * bsB;

    // ---- async stage: copy one 128x64 slab of A and B into LDS buffer `buf`
    auto stage_load = [&](int ks, int buf) {
        const int k0 = ks * KSLAB;
        // 128 rows * 64 cols = 1024 chunks of 8 bf16 (16B); 256 thr x 4 chunks
        #pragma unroll
        for (int c = 0; c < 4; ++c) {
            const int chunk = tid + c * 256;
            const int row   = chunk >> 3;          // 0..127
            const int cc    = (chunk & 7) << 3;    // 0,8,...,56
            const bf16_t* ga = Ab + (long)(tileM + row) * ldA + k0 + cc;
            const bf16_t* gb = Bb + (long)(tileN + row) * ldB + k0 + cc;
            const unsigned la =
                (unsigned)(uintptr_t)&smA[buf][row * LDSROW + cc];
            const unsigned lb =
                (unsigned)(uintptr_t)&smB[buf][row * LDSROW + cc];
            asm volatile("global_load_async_to_lds_b128 %0, %1, off"
                         :: "v"(la), "v"((unsigned long long)(uintptr_t)ga)
                         : "memory");
            asm volatile("global_load_async_to_lds_b128 %0, %1, off"
                         :: "v"(lb), "v"((unsigned long long)(uintptr_t)gb)
                         : "memory");
        }
    };

    // ---- register tile
    const int waveM = wave >> 2;                   // 0..1  -> 64 rows each
    const int waveN = wave & 3;                    // 0..3  -> 32 cols each
    const int mBase = waveM * 64;
    const int nBase = waveN * 32;
    const int rF     = lane & 15;
    const int kShift = (lane >> 4) << 3;           // 0 or 8 (ISA 16-bit frag layout)

    v8f acc[4][2];
    #pragma unroll
    for (int a = 0; a < 4; ++a)
        #pragma unroll
        for (int b = 0; b < 2; ++b)
            acc[a][b] = (v8f){};

    const int nk = K / KSLAB;

    stage_load(0, 0);
    asm volatile("s_wait_asynccnt 0x0" ::: "memory");
    __syncthreads();

    for (int ks = 0; ks < nk; ++ks) {
        const int buf = ks & 1;
        if (ks + 1 < nk)
            stage_load(ks + 1, buf ^ 1);           // streams while we compute

        const bf16_t* As = &smA[buf][0];
        const bf16_t* Bs = &smB[buf][0];

        #pragma unroll
        for (int kk = 0; kk < KSLAB; kk += 32) {
            Frag16 af[4], bfr[2];
            #pragma unroll
            for (int a = 0; a < 4; ++a) {
                const bf16_t* p = As + (mBase + a * 16 + rF) * LDSROW + kk + kShift;
                af[a].h[0] = *(const bf16x8*)(p);
                af[a].h[1] = *(const bf16x8*)(p + 16);
            }
            #pragma unroll
            for (int b = 0; b < 2; ++b) {
                const bf16_t* p = Bs + (nBase + b * 16 + rF) * LDSROW + kk + kShift;
                bfr[b].h[0] = *(const bf16x8*)(p);
                bfr[b].h[1] = *(const bf16x8*)(p + 16);
            }
            #pragma unroll
            for (int a = 0; a < 4; ++a)
                #pragma unroll
                for (int b = 0; b < 2; ++b)
                    acc[a][b] = __builtin_amdgcn_wmma_f32_16x16x32_bf16(
                        false, af[a].v, false, bfr[b].v,
                        (short)0, acc[a][b], false, false);
        }

        asm volatile("s_wait_asynccnt 0x0" ::: "memory");
        __syncthreads();
    }

    // ---- epilogue: C fragment layout: VGPR r, lane l -> row r+8*(l>>4), col l&15
    const int rowSub  = (lane >> 4) << 3;
    const int colSub  = lane & 15;
    float*  Cf = (float*)C;
    bf16_t* Cb = (bf16_t*)C;
    const long cbase = (long)z * bsC;

    #pragma unroll
    for (int a = 0; a < 4; ++a) {
        const int rowBase = tileM + mBase + a * 16 + rowSub;
        #pragma unroll
        for (int b = 0; b < 2; ++b) {
            const int col = tileN + nBase + b * 16 + colSub;
            const float bv = bias ? bias[col] : 0.0f;
            #pragma unroll
            for (int r = 0; r < 8; ++r) {
                const long idx = cbase + (long)(rowBase + r) * sCm + (long)col * sCn;
                const float val = acc[a][b][r] + bv;
                if (c_is_bf16) Cb[idx] = (bf16_t)val;
                else           Cf[idx] = val;
            }
        }
    }
}

// ---------------------------------------------------------------------------
// Row softmax (fp32 in, bf16 out). One 256-thread block per row of `ncols`.
// ---------------------------------------------------------------------------
__global__ __launch_bounds__(256)
void softmax_rows_bf16(const float* __restrict__ in, bf16_t* __restrict__ out, int ncols)
{
    __shared__ float red[256];
    const long row = blockIdx.x;
    const float* x = in  + row * (long)ncols;
    bf16_t*      y = out + row * (long)ncols;
    const int tid = threadIdx.x;

    float m = -3.402823466e38f;
    for (int j = tid; j < ncols; j += 256) m = fmaxf(m, x[j]);
    red[tid] = m;
    __syncthreads();
    for (int s = 128; s > 0; s >>= 1) {
        if (tid < s) red[tid] = fmaxf(red[tid], red[tid + s]);
        __syncthreads();
    }
    m = red[0];
    __syncthreads();

    float sum = 0.0f;
    for (int j = tid; j < ncols; j += 256) sum += __expf(x[j] - m);
    red[tid] = sum;
    __syncthreads();
    for (int s = 128; s > 0; s >>= 1) {
        if (tid < s) red[tid] += red[tid + s];
        __syncthreads();
    }
    const float inv = 1.0f / red[0];

    for (int j = tid; j < ncols; j += 256)
        y[j] = (bf16_t)(__expf(x[j] - m) * inv);
}

// ---------------------------------------------------------------------------
// Host-side orchestration
// ---------------------------------------------------------------------------
extern "C" void kernel_launch(void* const* d_in, const int* in_sizes, int n_in,
                              void* d_out, int out_size, void* d_ws, size_t ws_size,
                              hipStream_t stream)
{
    (void)in_sizes; (void)n_in; (void)out_size; (void)ws_size;

    constexpr int  Bn = 4, Sn = 1024, Dn = 1024;
    constexpr long BSD = (long)Bn * Sn * Dn;      // 4M elements
    constexpr long SD  = (long)Sn * Dn;           // 1M elements (per-batch matrix)
    constexpr long DD  = (long)Dn * Dn;           // 1M elements (weights)

    const float* x_q  = (const float*)d_in[0];
    const float* x_k  = (const float*)d_in[1];
    const float* x_v  = (const float*)d_in[2];
    const float* wq_w = (const float*)d_in[3];
    const float* wq_b = (const float*)d_in[4];
    const float* wk_w = (const float*)d_in[5];
    const float* wk_b = (const float*)d_in[6];
    const float* wv_w = (const float*)d_in[7];
    const float* wv_b = (const float*)d_in[8];
    const float* wo_w = (const float*)d_in[9];
    const float* wo_b = (const float*)d_in[10];

    // Workspace carve-up (256B aligned slabs), ~112 MB total
    char*  ws  = (char*)d_ws;
    size_t off = 0;
    auto take = [&](size_t bytes) -> char* {
        char* p = ws + off;
        off += (bytes + 255) & ~(size_t)255;
        return p;
    };
    bf16_t* xq_bf   = (bf16_t*)take(BSD * sizeof(bf16_t));
    bf16_t* xk_bf   = (bf16_t*)take(BSD * sizeof(bf16_t));
    bf16_t* xv_bf   = (bf16_t*)take(BSD * sizeof(bf16_t));
    bf16_t* wqw_bf  = (bf16_t*)take(DD  * sizeof(bf16_t));
    bf16_t* wkw_bf  = (bf16_t*)take(DD  * sizeof(bf16_t));
    bf16_t* wvw_bf  = (bf16_t*)take(DD  * sizeof(bf16_t));
    bf16_t* wow_bf  = (bf16_t*)take(DD  * sizeof(bf16_t));
    bf16_t* qT_bf   = (bf16_t*)take(BSD * sizeof(bf16_t));   // [b][i(KD), s(S)]
    bf16_t* k_bf    = (bf16_t*)take(BSD * sizeof(bf16_t));   // [b][s(S), j(KD)]
    bf16_t* v_bf    = (bf16_t*)take(BSD * sizeof(bf16_t));   // [b][t(S), j(KD)]
    float*  attn_f  = (float*) take(BSD * sizeof(float));    // [a][i, j]
    bf16_t* attn_bf = (bf16_t*)take(BSD * sizeof(bf16_t));   // [a][i, j]
    bf16_t* outT_bf = (bf16_t*)take((long)16 * SD * sizeof(bf16_t)); // [ab][t, i]

    // 1) fp32 -> bf16 conversions for GEMM operands
    {
        dim3 cb(256), cg(2048);
        cvt_f32_to_bf16<<<cg, cb, 0, stream>>>(x_q,  xq_bf,  BSD);
        cvt_f32_to_bf16<<<cg, cb, 0, stream>>>(x_k,  xk_bf,  BSD);
        cvt_f32_to_bf16<<<cg, cb, 0, stream>>>(x_v,  xv_bf,  BSD);
        cvt_f32_to_bf16<<<cg, cb, 0, stream>>>(wq_w, wqw_bf, DD);
        cvt_f32_to_bf16<<<cg, cb, 0, stream>>>(wk_w, wkw_bf, DD);
        cvt_f32_to_bf16<<<cg, cb, 0, stream>>>(wv_w, wvw_bf, DD);
        cvt_f32_to_bf16<<<cg, cb, 0, stream>>>(wo_w, wow_bf, DD);
    }

    const dim3 blk(32, 8, 1);
    auto gemm = [&](const bf16_t* A, long bsA, int divA, long ldA,
                    const bf16_t* Bp, long bsB, int modB, long ldB,
                    void* Cp, long bsC, long sCm, long sCn, int cbf,
                    const float* bias, int M, int N, int K, int batches) {
        dim3 grd(N / TILE_MN, M / TILE_MN, batches);
        gemm_bf16_wmma<<<grd, blk, 0, stream>>>(A, bsA, divA, ldA,
                                                Bp, bsB, modB, ldB,
                                                Cp, bsC, sCm, sCn, cbf,
                                                bias, M, N, K);
    };

    // 2) Projections (bias fused). Per batch: M=S rows, N=KD features, K=in-dim.
    //    q written TRANSPOSED: qT[b][i,s]  (sCm=1 over s, sCn=S over i)
    gemm(xq_bf, SD, 1, Dn, wqw_bf, 0, 1, Dn,
         qT_bf, SD, /*sCm=*/1, /*sCn=*/Sn, /*bf16*/1, wq_b, Sn, Dn, Dn, Bn);
    //    k, v written in natural [s, feature] layout
    gemm(xk_bf, SD, 1, Dn, wkw_bf, 0, 1, Dn,
         k_bf,  SD, /*sCm=*/Dn, /*sCn=*/1, 1, wk_b, Sn, Dn, Dn, Bn);
    gemm(xv_bf, SD, 1, Dn, wvw_bf, 0, 1, Dn,
         v_bf,  SD, /*sCm=*/Dn, /*sCn=*/1, 1, wv_b, Sn, Dn, Dn, Bn);

    // 3) attn[b,i,j] = sum_s qT[b][i,s] * k_bf[b][j,s]  -> fp32 for softmax
    gemm(qT_bf, SD, 1, Sn, k_bf, SD, 4, Dn,
         attn_f, SD, /*sCm=*/Dn, /*sCn=*/1, /*fp32*/0, nullptr, Dn, Sn, Sn, Bn);

    // 4) softmax over j, emit bf16
    softmax_rows_bf16<<<dim3(Bn * Dn), dim3(256), 0, stream>>>(attn_f, attn_bf, Sn);

    // 5) out[ab][i,t] = sum_j attn[a][i,j] * v[bb][t,j] ; a = z/4, bb = z%4
    //    stored transposed: outT[ab][t,i] (sCm=1 over i, sCn=Dn over t)
    gemm(attn_bf, SD, /*divA=*/4, Sn, v_bf, SD, /*modB=*/4, Dn,
         outT_bf, SD, /*sCm=*/1, /*sCn=*/Dn, 1, nullptr, Dn, Sn, Sn, Bn * Bn);

    // 6) y[r,n] = sum_m outT_flat[r,m] * wo_w[n,m] + wo_b[n], r = ab*S + t
    gemm(outT_bf, 0, 1, Dn, wow_bf, 0, 1, Dn,
         d_out, 0, /*sCm=*/Dn, /*sCn=*/1, /*fp32*/0, wo_b,
         Bn * Bn * Sn, Dn, Dn, 1);
}
